// PPI_GAT_29850022707671
// MI455X (gfx1250) — compile-verified, matrix operand order checked by reading, not run on hardware
//
#include <hip/hip_runtime.h>
#include <hip/hip_bf16.h>

// ---------------- problem constants (match reference) ----------------
constexpr int NN   = 50000;
constexpr int EE   = 800000;
constexpr int ETOT = EE + NN;      // edges + self loops = 850000
constexpr int INF_ = 128;
constexpr int H1_  = 4;
constexpr int HID_ = 64;
constexpr int F1   = H1_ * HID_;   // 256
constexpr int H2_  = 6;
constexpr int OUT_ = 121;
constexpr int F3   = H2_ * OUT_;   // 726
constexpr int F3P  = 736;          // padded to multiple of 32 for WMMA tiles
constexpr float NEG_SLOPE = 0.2f;

typedef __attribute__((ext_vector_type(16))) _Float16 v16h;
typedef __attribute__((ext_vector_type(8)))  _Float16 v8h;
typedef __attribute__((ext_vector_type(8)))  float    v8f;

// ---------------- operand packing ----------------
// fp32 -> fp16, flat copy (activations, row-major, lda == K)
__global__ void gat_pack_f16(const float* __restrict__ A, _Float16* __restrict__ O, size_t n) {
  size_t i = (size_t)blockIdx.x * blockDim.x + threadIdx.x;
  if (i < n) O[i] = (_Float16)A[i];
}

// weights W[K x Nreal] (row-major) -> fp16 transposed, zero-padded: O[n*K+k], n < Npad
__global__ void gat_pack_wT_f16(const float* __restrict__ W, _Float16* __restrict__ O,
                                int K, int Nreal, int Npad) {
  int t = blockIdx.x * blockDim.x + threadIdx.x;
  if (t >= K * Npad) return;
  const int n = t / K, k = t % K;
  O[t] = (n < Nreal) ? (_Float16)W[(size_t)k * Nreal + n] : (_Float16)0.f;
}

// ---------------- WMMA GEMM: C[M x Npad] = A16[M x K] * B16T[Npad x K]^T --------
// A16 fp16 row-major; B16T fp16 transposed (row n holds column n of W), zero-padded.
// Each wave computes a 16x32 tile: 6x global_load_b128 + 2x v_wmma per k-step.
__global__ __launch_bounds__(128)
void gat_gemm_wmma(const _Float16* __restrict__ A16, const _Float16* __restrict__ B16T,
                   float* __restrict__ C, int M, int K, int ldc) {
  const int lane = threadIdx.x & 31;
  const int wave = threadIdx.x >> 5;
  const int half = lane >> 4;
  const int r    = lane & 15;
  const int m0   = (blockIdx.x * 4 + wave) * 16;
  const int n0   = blockIdx.y * 32;
  if (m0 >= M) return;

  const _Float16* arow = A16  + (size_t)(m0 + r) * K;
  const _Float16* br0  = B16T + (size_t)(n0 + r) * K;
  const _Float16* br1  = B16T + (size_t)(n0 + 16 + r) * K;

  v8f acc0 = {};
  v8f acc1 = {};

  for (int k0 = 0; k0 < K; k0 += 32) {
    __builtin_prefetch(arow + k0 + 64, 0, 0);   // global_prefetch_b8
    // 16-bit A-matrix 16x32 layout (ISA 7.12.2): frag halves are two contiguous
    // 8-element runs at k0 + 8*half and k0 + 16 + 8*half -> 16B-aligned b128 loads.
    const v8h alo  = *(const v8h*)(arow + k0 + 8 * half);
    const v8h ahi  = *(const v8h*)(arow + k0 + 16 + 8 * half);
    const v8h b0lo = *(const v8h*)(br0  + k0 + 8 * half);
    const v8h b0hi = *(const v8h*)(br0  + k0 + 16 + 8 * half);
    const v8h b1lo = *(const v8h*)(br1  + k0 + 8 * half);
    const v8h b1hi = *(const v8h*)(br1  + k0 + 16 + 8 * half);
    const v16h af  = __builtin_shufflevector(alo, ahi, 0,1,2,3,4,5,6,7,8,9,10,11,12,13,14,15);
    const v16h bf0 = __builtin_shufflevector(b0lo, b0hi, 0,1,2,3,4,5,6,7,8,9,10,11,12,13,14,15);
    const v16h bf1 = __builtin_shufflevector(b1lo, b1hi, 0,1,2,3,4,5,6,7,8,9,10,11,12,13,14,15);
    acc0 = __builtin_amdgcn_wmma_f32_16x16x32_f16(false, af, false, bf0,
                                                  (short)0, acc0, false, false);
    acc1 = __builtin_amdgcn_wmma_f32_16x16x32_f16(false, af, false, bf1,
                                                  (short)0, acc1, false, false);
  }

  // C/D layout: lane -> col = lane&15 ; VGPR i -> row = i + 8*(lane>>4).
  // ldc is padded, so stores are unconditional.
#pragma unroll
  for (int i = 0; i < 8; ++i) {
    float* crow = C + (size_t)(m0 + i + 8 * half) * ldc;
    crow[n0 + r]      = acc0[i];
    crow[n0 + 16 + r] = acc1[i];
  }
}

// ---------------- per-node attention logits ----------------
__global__ void gat_attn_logits(const float* __restrict__ h,
                                const float* __restrict__ a_src,
                                const float* __restrict__ a_dst,
                                float* __restrict__ als, float* __restrict__ ald,
                                int H, int C, int ldh) {
  int t = blockIdx.x * blockDim.x + threadIdx.x;
  if (t >= NN * H) return;
  const int n = t / H, hh = t % H;
  const float* row = h + (size_t)n * ldh + hh * C;
  const float* as  = a_src + hh * C;
  const float* ad  = a_dst + hh * C;
  float s = 0.f, d = 0.f;
  for (int c = 0; c < C; ++c) { float v = row[c]; s += v * as[c]; d += v * ad[c]; }
  als[t] = s; ald[t] = d;
}

// ---------------- fill ----------------
__global__ void gat_fill(float* __restrict__ p, float v, size_t n) {
  size_t i = (size_t)blockIdx.x * blockDim.x + threadIdx.x;
  if (i < n) p[i] = v;
}

// ---------------- edge helpers ----------------
__device__ inline void edge_nodes(int e, const int* __restrict__ src,
                                  const int* __restrict__ dst, int& s, int& d) {
  if (e < EE) { s = src[e]; d = dst[e]; } else { s = e - EE; d = e - EE; }
}

__device__ inline void atomicMaxF(float* addr, float v) {
  if (v >= 0.f) atomicMax((int*)addr, __float_as_int(v));
  else          atomicMin((unsigned int*)addr, __float_as_uint(v));
}

// pass 1: leaky-relu logits + segment max
__global__ void gat_edge_max(const int* __restrict__ src, const int* __restrict__ dst,
                             const float* __restrict__ als, const float* __restrict__ ald,
                             float* __restrict__ ew, float* __restrict__ mx, int H) {
  int t = blockIdx.x * blockDim.x + threadIdx.x;
  if (t >= ETOT * H) return;
  const int e = t / H, hh = t % H;
  int s, d; edge_nodes(e, src, dst, s, d);
  float v = als[s * H + hh] + ald[d * H + hh];
  v = (v > 0.f) ? v : NEG_SLOPE * v;
  ew[t] = v;
  atomicMaxF(&mx[d * H + hh], v);
}

// pass 2: exp + segment sum
__global__ void gat_edge_expsum(const int* __restrict__ src, const int* __restrict__ dst,
                                float* __restrict__ ew, const float* __restrict__ mx,
                                float* __restrict__ sm, int H) {
  int t = blockIdx.x * blockDim.x + threadIdx.x;
  if (t >= ETOT * H) return;
  const int e = t / H, hh = t % H;
  int s, d; edge_nodes(e, src, dst, s, d);
  float v = __expf(ew[t] - mx[d * H + hh]);
  ew[t] = v;
  atomicAdd(&sm[d * H + hh], v);
}

// pass 3: weighted scatter of messages
__global__ void gat_edge_scatter(const int* __restrict__ src, const int* __restrict__ dst,
                                 const float* __restrict__ h, const float* __restrict__ ew,
                                 const float* __restrict__ sm, float* __restrict__ acc,
                                 int H, int C, int ldh) {
  int t = blockIdx.x * blockDim.x + threadIdx.x;
  if (t >= ETOT * H) return;
  const int e = t / H, hh = t % H;
  int s, d; edge_nodes(e, src, dst, s, d);
  const float alpha = ew[t] / (sm[d * H + hh] + 1e-16f);
  const float* hrow = h   + (size_t)s * ldh + hh * C;
  float*       arow = acc + (size_t)d * ldh + hh * C;
  if ((C & 3) == 0) {
    for (int c = 0; c < C; c += 4) {
      const float4 hv = *(const float4*)(hrow + c);
      atomicAdd(arow + c + 0, alpha * hv.x);
      atomicAdd(arow + c + 1, alpha * hv.y);
      atomicAdd(arow + c + 2, alpha * hv.z);
      atomicAdd(arow + c + 3, alpha * hv.w);
    }
  } else {
    for (int c = 0; c < C; ++c) atomicAdd(arow + c, alpha * hrow[c]);
  }
}

// ---------------- epilogues ----------------
// layers 1/2: x_next(fp16) = elu(acc + b [+ res])  (consumed only as GEMM A-operand)
__global__ void gat_epilogue_elu(const float* __restrict__ acc, const float* __restrict__ b,
                                 const float* __restrict__ res, _Float16* __restrict__ out16) {
  int t = blockIdx.x * blockDim.x + threadIdx.x;
  if (t >= NN * F1) return;
  const int f = t % F1;
  float v = acc[t] + b[f];
  if (res) v += res[t];
  v = (v > 0.f) ? v : (__expf(v) - 1.0f);
  out16[t] = (_Float16)v;
}

// layer 3: out = sigmoid(mean_heads(acc) + b)
__global__ void gat_epilogue_out(const float* __restrict__ acc, const float* __restrict__ b,
                                 float* __restrict__ out) {
  int t = blockIdx.x * blockDim.x + threadIdx.x;
  if (t >= NN * OUT_) return;
  const int n = t / OUT_, o = t % OUT_;
  float v = 0.f;
#pragma unroll
  for (int hh = 0; hh < H2_; ++hh) v += acc[(size_t)n * F3P + hh * OUT_ + o];
  v = v * (1.0f / (float)H2_) + b[o];
  out[t] = 1.0f / (1.0f + __expf(-v));
}

// ---------------- launcher ----------------
static inline void launch_gemm(const _Float16* A16, const _Float16* B16T, float* C,
                               int M, int K, int ldc, hipStream_t stream) {
  dim3 block(128);
  dim3 grid((M / 16 + 3) / 4, ldc / 32);
  gat_gemm_wmma<<<grid, block, 0, stream>>>(A16, B16T, C, M, K, ldc);
}

extern "C" void kernel_launch(void* const* d_in, const int* in_sizes, int n_in,
                              void* d_out, int out_size, void* d_ws, size_t ws_size,
                              hipStream_t stream) {
  const float* x      = (const float*)d_in[0];
  const int*   src    = (const int*)  d_in[1];
  const int*   dst    = (const int*)  d_in[2];
  const float* W1     = (const float*)d_in[3];
  const float* a1s    = (const float*)d_in[4];
  const float* a1d    = (const float*)d_in[5];
  const float* b1     = (const float*)d_in[6];
  const float* W2     = (const float*)d_in[7];
  const float* a2s    = (const float*)d_in[8];
  const float* a2d    = (const float*)d_in[9];
  const float* b2     = (const float*)d_in[10];
  const float* Wres2  = (const float*)d_in[11];
  const float* W3     = (const float*)d_in[12];
  const float* a3s    = (const float*)d_in[13];
  const float* a3d    = (const float*)d_in[14];
  const float* b3     = (const float*)d_in[15];
  float*       out    = (float*)d_out;

  // workspace layout
  float* WS     = (float*)d_ws;
  float* bufH   = WS;                          // NN*F3P  transformed features (fp32)
  float* bufAcc = bufH   + (size_t)NN * F3P;   // NN*F3P  attention aggregate
  float* bufRes = bufAcc + (size_t)NN * F3P;   // NN*F1   residual
  float* bufAls = bufRes + (size_t)NN * F1;    // NN*8
  float* bufAld = bufAls + (size_t)NN * 8;     // NN*8
  float* bufMx  = bufAld + (size_t)NN * 8;     // NN*8
  float* bufSm  = bufMx  + (size_t)NN * 8;     // NN*8
  float* bufEw  = bufSm  + (size_t)NN * 8;     // ETOT*6
  _Float16* A16x  = (_Float16*)(bufEw + (size_t)ETOT * 6); // NN*128 fp16 input
  _Float16* A16x2 = A16x  + (size_t)NN * INF_;             // NN*256 fp16 activations
  _Float16* W1t   = A16x2 + (size_t)NN * F1;               // 256*128
  _Float16* W2t   = W1t   + (size_t)F1 * INF_;             // 256*256
  _Float16* Wrt   = W2t   + (size_t)F1 * F1;               // 256*256
  _Float16* W3t   = Wrt   + (size_t)F1 * F1;               // 736*256

  const int TB = 256;
  auto blocks = [](size_t n, int tb) { return (unsigned)((n + tb - 1) / tb); };

  // ---------------- operand packing (fp16) ----------------
  gat_pack_f16<<<blocks((size_t)NN * INF_, TB), TB, 0, stream>>>(x, A16x, (size_t)NN * INF_);
  gat_pack_wT_f16<<<blocks((size_t)INF_ * F1, TB), TB, 0, stream>>>(W1, W1t, INF_, F1, F1);
  gat_pack_wT_f16<<<blocks((size_t)F1 * F1, TB), TB, 0, stream>>>(W2, W2t, F1, F1, F1);
  gat_pack_wT_f16<<<blocks((size_t)F1 * F1, TB), TB, 0, stream>>>(Wres2, Wrt, F1, F1, F1);
  gat_pack_wT_f16<<<blocks((size_t)F1 * F3P, TB), TB, 0, stream>>>(W3, W3t, F1, F3, F3P);

  // ---------------- layer 1 ----------------
  launch_gemm(A16x, W1t, bufH, NN, INF_, F1, stream);
  gat_attn_logits<<<blocks((size_t)NN * H1_, TB), TB, 0, stream>>>(bufH, a1s, a1d, bufAls, bufAld, H1_, HID_, F1);
  gat_fill<<<blocks((size_t)NN * H1_, TB), TB, 0, stream>>>(bufMx, -3.0e38f, (size_t)NN * H1_);
  gat_fill<<<blocks((size_t)NN * H1_, TB), TB, 0, stream>>>(bufSm, 0.f, (size_t)NN * H1_);
  gat_fill<<<blocks((size_t)NN * F1, TB), TB, 0, stream>>>(bufAcc, 0.f, (size_t)NN * F1);
  gat_edge_max<<<blocks((size_t)ETOT * H1_, TB), TB, 0, stream>>>(src, dst, bufAls, bufAld, bufEw, bufMx, H1_);
  gat_edge_expsum<<<blocks((size_t)ETOT * H1_, TB), TB, 0, stream>>>(src, dst, bufEw, bufMx, bufSm, H1_);
  gat_edge_scatter<<<blocks((size_t)ETOT * H1_, TB), TB, 0, stream>>>(src, dst, bufH, bufEw, bufSm, bufAcc, H1_, HID_, F1);
  gat_epilogue_elu<<<blocks((size_t)NN * F1, TB), TB, 0, stream>>>(bufAcc, b1, nullptr, A16x2);

  // ---------------- layer 2 (residual) ----------------
  launch_gemm(A16x2, W2t, bufH, NN, F1, F1, stream);
  launch_gemm(A16x2, Wrt, bufRes, NN, F1, F1, stream);
  gat_attn_logits<<<blocks((size_t)NN * H1_, TB), TB, 0, stream>>>(bufH, a2s, a2d, bufAls, bufAld, H1_, HID_, F1);
  gat_fill<<<blocks((size_t)NN * H1_, TB), TB, 0, stream>>>(bufMx, -3.0e38f, (size_t)NN * H1_);
  gat_fill<<<blocks((size_t)NN * H1_, TB), TB, 0, stream>>>(bufSm, 0.f, (size_t)NN * H1_);
  gat_fill<<<blocks((size_t)NN * F1, TB), TB, 0, stream>>>(bufAcc, 0.f, (size_t)NN * F1);
  gat_edge_max<<<blocks((size_t)ETOT * H1_, TB), TB, 0, stream>>>(src, dst, bufAls, bufAld, bufEw, bufMx, H1_);
  gat_edge_expsum<<<blocks((size_t)ETOT * H1_, TB), TB, 0, stream>>>(src, dst, bufEw, bufMx, bufSm, H1_);
  gat_edge_scatter<<<blocks((size_t)ETOT * H1_, TB), TB, 0, stream>>>(src, dst, bufH, bufEw, bufSm, bufAcc, H1_, HID_, F1);
  gat_epilogue_elu<<<blocks((size_t)NN * F1, TB), TB, 0, stream>>>(bufAcc, b2, bufRes, A16x2);

  // ---------------- layer 3 (mean heads + sigmoid) ----------------
  launch_gemm(A16x2, W3t, bufH, NN, F1, F3P, stream);
  gat_attn_logits<<<blocks((size_t)NN * H2_, TB), TB, 0, stream>>>(bufH, a3s, a3d, bufAls, bufAld, H2_, OUT_, F3P);
  gat_fill<<<blocks((size_t)NN * H2_, TB), TB, 0, stream>>>(bufMx, -3.0e38f, (size_t)NN * H2_);
  gat_fill<<<blocks((size_t)NN * H2_, TB), TB, 0, stream>>>(bufSm, 0.f, (size_t)NN * H2_);
  gat_fill<<<blocks((size_t)NN * F3P, TB), TB, 0, stream>>>(bufAcc, 0.f, (size_t)NN * F3P);
  gat_edge_max<<<blocks((size_t)ETOT * H2_, TB), TB, 0, stream>>>(src, dst, bufAls, bufAld, bufEw, bufMx, H2_);
  gat_edge_expsum<<<blocks((size_t)ETOT * H2_, TB), TB, 0, stream>>>(src, dst, bufEw, bufMx, bufSm, H2_);
  gat_edge_scatter<<<blocks((size_t)ETOT * H2_, TB), TB, 0, stream>>>(src, dst, bufH, bufEw, bufSm, bufAcc, H2_, OUT_, F3P);
  gat_epilogue_out<<<blocks((size_t)NN * OUT_, TB), TB, 0, stream>>>(bufAcc, b3, out);
}
